// fMRI3DGNN_16595753631817
// MI455X (gfx1250) — matrix operand order repaired and checked by reading, not compiled, not run on hardware
//
#include <hip/hip_runtime.h>
#include <stdint.h>
#include <math.h>

// ---------------------------------------------------------------------------
// fMRI 3D GNN forward for MI455X (gfx1250, wave32, WMMA).
//
// Roofline:
//  * GEMM1 streams 640 MB of f32 weights -> HBM floor ~27.5us @ 23.3 TB/s.
//    bf16 WMMA via LDS-tiled GEMM keeps it bandwidth bound (coalesced global
//    loads with immediate offsets, b128 LDS fragment loads, 4-8 wmma / K-step).
//  * GAT linears (~80 GFLOP) use the wide-tile variant (8 wmma / K-step).
//  * Edge softmax/aggregation gathers from L2-resident activations (52 MB);
//    CSR-by-dst (in-degree <= 200), no float atomics, deterministic.
// ---------------------------------------------------------------------------

typedef __attribute__((ext_vector_type(16))) __bf16 v16bf;
typedef __attribute__((ext_vector_type(8)))  __bf16 v8bf;
typedef __attribute__((ext_vector_type(8)))  float  v8f;

#define DEV __device__ __forceinline__

constexpr int BB    = 64;           // batch
constexpr int NN    = 200;          // nodes per graph
constexpr int NT    = BB * NN;      // 12800 total nodes
constexpr int ADJN  = NN * NN;      // 40000 adjacency entries / batch
constexpr int EPER  = ADJN / 4;     // 10000 edges / batch
constexpr int DIN   = 4000;
constexpr int TIECAP = 64;

DEV __bf16 f2bf(float f) {
    union { float f; uint32_t u; } a; a.f = f;
    uint32_t u = a.u;
    u += 0x7FFFu + ((u >> 16) & 1u);             // round-to-nearest-even
    union { unsigned short s; __bf16 b; } r;
    r.s = (unsigned short)(u >> 16);
    return r.b;
}
DEV unsigned pack2bf(float f0, float f1) {
    union { float f; uint32_t u; } a0, a1; a0.f = f0; a1.f = f1;
    uint32_t u0 = a0.u; u0 += 0x7FFFu + ((u0 >> 16) & 1u);
    uint32_t u1 = a1.u; u1 += 0x7FFFu + ((u1 >> 16) & 1u);
    return (u0 >> 16) | (u1 & 0xFFFF0000u);
}
DEV float geluf(float x) { return 0.5f * x * (1.0f + erff(x * 0.70710678118654752f)); }

union frag16 { v16bf v; v8bf h[2]; };

// ---------------------------------------------------------------------------
// LDS-tiled bf16 WMMA GEMM: C[M,N] = act(A[M,K](bf16) @ B[K,N](f32->bf16) + bias)
// N, K compile-time so all strided B loads / C stores fold into immediate
// VMEM offsets (per-k byte stride 4*N fits the signed 24-bit ioffset).
// Block: 256 threads (8 waves, 2x4 wave grid), tile BM=64 x BN=64*NF x BK=32.
// Each wave owns 2 x NF 16x16 tiles -> 2*NF wmma per K-step. A fragment =
// two ds_load_b128 from row-major sA; B fragment = two ds_load_b128 from
// transposed sB[n][k]. M % 64 == 0, K % 32 == 0. N ragged (clamped loads +
// cndmask zero; store guards only -- EXEC all-ones at every wmma).
// ACT: 0 = none, 1 = sigmoid.
// ---------------------------------------------------------------------------
template <int NF, int N, int K, int ACT>
__global__ void __launch_bounds__(256)
wmma_gemm_bf16_kernel(const __bf16* __restrict__ A, const float* __restrict__ B,
                      const float* __restrict__ bias, float* __restrict__ C, int M)
{
    constexpr int BN  = 64 * NF;            // 128 (NF=2) or 256 (NF=4)
    constexpr int TPC = 256 / BN;           // threads per column: 2 or 1
    constexpr int KPT = 16 / TPC;           // k-pairs per thread: 8 or 16
    constexpr int NTB = (N + BN - 1) / BN;  // N tiles

    __shared__ __align__(16) __bf16 sA[64 * 32];    // [m][k]
    __shared__ __align__(16) __bf16 sB[BN * 32];    // [n][k] (transposed)

    int bm  = blockIdx.x / NTB;
    int bn  = blockIdx.x % NTB;
    int tm  = bm << 6;
    int tn  = bn * BN;

    int tid  = threadIdx.x;
    int lane = tid & 31;
    int wid  = tid >> 5;            // 0..7
    int wm   = wid & 1;             // wave M subtile (2 x 32 rows)
    int wn   = wid >> 1;            // wave N subtile (4 x 16*NF cols)
    bool hi  = lane >= 16;
    int  l15 = lane & 15;

    // cooperative-load thread mappings (fixed per thread)
    int arow = tid >> 2, aq = tid & 3;      // A: 64 rows x 4 x (8 halves)
    int bnid = tid % BN;                    // B column within tile
    int kpb  = (tid / BN) * KPT;            // first k-pair for this thread

    int  gn  = tn + bnid;
    bool inb = gn < N;
    int  gnc = inb ? gn : (N - 1);          // clamped: loads always legal

    v8f c[2][NF];
#pragma unroll
    for (int sm = 0; sm < 2; sm++)
#pragma unroll
        for (int sn = 0; sn < NF; sn++) c[sm][sn] = v8f{};

    for (int k0 = 0; k0 < K; k0 += 32) {
        __syncthreads();
        // ---- stage A tile (bf16, b128 loads) ----
        {
            const uint4* src = (const uint4*)(A + (size_t)(tm + arow) * K + k0 + aq * 8);
            *(uint4*)&sA[arow * 32 + aq * 8] = *src;
        }
        // ---- stage B tile transposed (coalesced loads, immediate k offsets) ----
        {
            const float* Bp = B + (size_t)(k0 + kpb * 2) * N + gnc;
            unsigned tmp[KPT];
#pragma unroll
            for (int i = 0; i < KPT; i++) {
                float f0 = Bp[(size_t)(2 * i) * N];
                float f1 = Bp[(size_t)(2 * i + 1) * N];
                unsigned pk = pack2bf(f0, f1);
                tmp[i] = inb ? pk : 0u;
            }
#pragma unroll
            for (int i = 0; i < KPT; i += 4)
                *(uint4*)&sB[bnid * 32 + (kpb + i) * 2] =
                    make_uint4(tmp[i], tmp[i + 1], tmp[i + 2], tmp[i + 3]);
            if (k0 + 32 < K)
                __builtin_prefetch(&Bp[(size_t)32 * N], 0, 1);
        }
        __syncthreads();

        // ---- fragments from LDS (two b128 each) ----
        frag16 fa[2], fb[NF];
#pragma unroll
        for (int sm = 0; sm < 2; sm++) {
            int ml = wm * 32 + sm * 16 + l15;
            fa[sm].h[0] = *(const v8bf*)&sA[ml * 32 + (hi ? 8 : 0)];
            fa[sm].h[1] = *(const v8bf*)&sA[ml * 32 + (hi ? 24 : 16)];
        }
#pragma unroll
        for (int sn = 0; sn < NF; sn++) {
            int nl = wn * (16 * NF) + sn * 16 + l15;
            fb[sn].h[0] = *(const v8bf*)&sB[nl * 32 + (hi ? 8 : 0)];
            fb[sn].h[1] = *(const v8bf*)&sB[nl * 32 + (hi ? 24 : 16)];
        }
        // ---- 2*NF WMMAs ----
#pragma unroll
        for (int sm = 0; sm < 2; sm++)
#pragma unroll
            for (int sn = 0; sn < NF; sn++)
                c[sm][sn] = __builtin_amdgcn_wmma_f32_16x16x32_bf16(
                    false, fa[sm].v, false, fb[sn].v, (short)0, c[sm][sn], false, false);
    }

    // ---- epilogue: C/D layout: VGPR r -> row r (+8 for lanes>=16), col lane&15 ----
#pragma unroll
    for (int sm = 0; sm < 2; sm++) {
#pragma unroll
        for (int sn = 0; sn < NF; sn++) {
            int n = tn + wn * (16 * NF) + sn * 16 + l15;
            if (n >= N) continue;
            float bv = bias[n];
            int m0 = tm + wm * 32 + sm * 16 + (hi ? 8 : 0);
            float* Cp = C + (size_t)m0 * N + n;
#pragma unroll
            for (int r = 0; r < 8; r++) {
                float v = c[sm][sn][r] + bv;
                if (ACT == 1) v = 1.0f / (1.0f + expf(-v));
                Cp[(size_t)r * N] = v;
            }
        }
    }
}

// ---------------------------------------------------------------------------
__global__ void cvt_bf16_kernel(const float* __restrict__ src, __bf16* __restrict__ dst, int n)
{
    int i = blockIdx.x * blockDim.x + threadIdx.x;
    if (i < n) dst[i] = f2bf(src[i]);
}

// Symmetrize adj and compute per-row mean / unbiased std. One wave per row.
__global__ void sym_stats_kernel(const float* __restrict__ adjRaw, float* __restrict__ adjSym,
                                 float* __restrict__ means, float* __restrict__ stds)
{
    int wid  = (blockIdx.x * blockDim.x + threadIdx.x) >> 5;
    int lane = threadIdx.x & 31;
    if (wid >= NT) return;
    int b = wid / NN, i = wid % NN;
    const float* Ab = adjRaw + (size_t)b * ADJN;
    float s1 = 0.f, s2 = 0.f;
    for (int j = lane; j < NN; j += 32) {
        float v = 0.5f * (Ab[i * NN + j] + Ab[j * NN + i]);
        adjSym[(size_t)b * ADJN + i * NN + j] = v;
        s1 += v; s2 += v * v;
    }
    for (int o = 16; o; o >>= 1) { s1 += __shfl_xor(s1, o, 32); s2 += __shfl_xor(s2, o, 32); }
    if (lane == 0) {
        float mean = s1 / (float)NN;
        float var  = (s2 - (float)NN * mean * mean) / (float)(NN - 1);
        if (var < 0.f) var = 0.f;
        means[wid] = mean;
        stds[wid]  = sqrtf(var) + 1e-6f;
    }
}

// Tiny MLP (2->8 gelu ->16) + LayerNorm(16). One thread per node.
__global__ void mlp_ln_kernel(const float* __restrict__ means, const float* __restrict__ stds,
                              const float* __restrict__ W1, const float* __restrict__ b1,
                              const float* __restrict__ W2, const float* __restrict__ b2,
                              const float* __restrict__ g,  const float* __restrict__ be,
                              float* __restrict__ x0)
{
    int n = blockIdx.x * blockDim.x + threadIdx.x;
    if (n >= NT) return;
    float in0 = means[n], in1 = stds[n];
    float h[8];
#pragma unroll
    for (int j = 0; j < 8; j++) h[j] = geluf(in0 * W1[j] + in1 * W1[8 + j] + b1[j]);
    float o[16]; float mu = 0.f;
#pragma unroll
    for (int j = 0; j < 16; j++) {
        float v = b2[j];
#pragma unroll
        for (int k = 0; k < 8; k++) v += h[k] * W2[k * 16 + j];
        o[j] = v; mu += v;
    }
    mu *= (1.f / 16.f);
    float var = 0.f;
#pragma unroll
    for (int j = 0; j < 16; j++) { float d = o[j] - mu; var += d * d; }
    var *= (1.f / 16.f);
    float inv = rsqrtf(var + 1e-5f);
#pragma unroll
    for (int j = 0; j < 16; j++) x0[n * 16 + j] = (o[j] - mu) * inv * g[j] + be[j];
}

// ---------------------------------------------------------------------------
// Top-k via two-level radix histogram on float bits (values > 0 -> monotonic).
// ---------------------------------------------------------------------------
__global__ void hist_hi_kernel(const float* __restrict__ adjSym, unsigned* __restrict__ hist)
{
    int idx = blockIdx.x * blockDim.x + threadIdx.x;
    if (idx >= BB * ADJN) return;
    int b = idx / ADJN;
    unsigned u = __float_as_uint(adjSym[idx]);
    atomicAdd(&hist[(size_t)b * 65536 + (u >> 16)], 1u);
}

__global__ void hist_lo_kernel(const float* __restrict__ adjSym, const int* __restrict__ hiBin,
                               unsigned* __restrict__ hist)
{
    int idx = blockIdx.x * blockDim.x + threadIdx.x;
    if (idx >= BB * ADJN) return;
    int b = idx / ADJN;
    unsigned u = __float_as_uint(adjSym[idx]);
    if ((int)(u >> 16) == hiBin[b])
        atomicAdd(&hist[(size_t)b * 65536 + (u & 0xFFFFu)], 1u);
}

// One block per batch: scan 65536 bins from the top, find crossing bin.
__global__ void hist_scan_kernel(const unsigned* __restrict__ hist, const int* __restrict__ needIn,
                                 int needConst, int* __restrict__ outBin, int* __restrict__ outNeed)
{
    int b = blockIdx.x;
    int t = threadIdx.x;                 // 256 threads
    const unsigned* H = hist + (size_t)b * 65536;
    __shared__ unsigned csum[256];
    unsigned s = 0;
    for (int i = 0; i < 256; i++) s += H[t * 256 + i];
    csum[t] = s;
    __syncthreads();
    if (t == 0) {
        int need = needIn ? needIn[b] : needConst;
        int cum = 0;
        int chunk = 255;
        for (; chunk > 0; chunk--) {
            if (cum + (int)csum[chunk] >= need) break;
            cum += (int)csum[chunk];
        }
        int bin = (chunk << 8) | 255;
        for (; bin > (chunk << 8); bin--) {
            if (cum + (int)H[bin] >= need) break;
            cum += (int)H[bin];
        }
        outBin[b]  = bin;
        outNeed[b] = need - cum;   // #elements to take at exactly this bin
    }
}

// Record (in index order) flat indices equal to the exact threshold value.
__global__ void tie_scan_kernel(const float* __restrict__ adjSym, const int* __restrict__ hiBin,
                                const int* __restrict__ loBin, int* __restrict__ tieIdx,
                                int* __restrict__ tieCnt)
{
    int b = blockIdx.x * blockDim.x + threadIdx.x;
    if (b >= BB) return;
    unsigned thr = ((unsigned)hiBin[b] << 16) | (unsigned)loBin[b];
    const float* Ab = adjSym + (size_t)b * ADJN;
    int* tix = tieIdx + b * TIECAP;
    int c = 0;
    for (int i = 0; i < ADJN; i++) {
        if (__float_as_uint(Ab[i]) == thr) {
            if (c < TIECAP) tix[c] = i;
            c++;
        }
    }
    tieCnt[b] = c;
}

DEV int edge_selected(unsigned u, unsigned thr, int flat,
                      const int* __restrict__ tix, int take, int tc)
{
    if (u > thr) return 1;
    if (u != thr) return 0;
    int lim = tc < TIECAP ? tc : TIECAP;
    for (int t = 0; t < lim; t++)
        if (tix[t] == flat) return (t < take) ? 1 : 0;
    return 0;
}

// Per-dst in-degree (deterministic; one thread per destination node).
__global__ void deg_kernel(const float* __restrict__ adjSym, const int* __restrict__ hiBin,
                           const int* __restrict__ loBin, const int* __restrict__ tieNeed,
                           const int* __restrict__ tieCnt, const int* __restrict__ tieIdx,
                           int* __restrict__ deg)
{
    int d = blockIdx.x * blockDim.x + threadIdx.x;
    if (d >= NT) return;
    int b = d / NN, col = d % NN;
    unsigned thr = ((unsigned)hiBin[b] << 16) | (unsigned)loBin[b];
    const float* Ab = adjSym + (size_t)b * ADJN;
    const int* tix = tieIdx + b * TIECAP;
    int take = tieNeed[b], tc = tieCnt[b];
    int cnt = 0;
    for (int r = 0; r < NN; r++) {
        int flat = r * NN + col;
        cnt += edge_selected(__float_as_uint(Ab[flat]), thr, flat, tix, take, tc);
    }
    deg[d] = cnt;
}

// Exclusive prefix sum over 12800 degrees. Single block of 256 threads.
__global__ void prefix_kernel(const int* __restrict__ deg, int* __restrict__ off)
{
    __shared__ int part[256];
    __shared__ int base[256];
    int t = threadIdx.x;
    int s = 0;
    for (int i = t * 50; i < t * 50 + 50; i++) s += deg[i];
    part[t] = s;
    __syncthreads();
    if (t == 0) {
        int acc = 0;
        for (int i = 0; i < 256; i++) { base[i] = acc; acc += part[i]; }
        off[NT] = acc;
    }
    __syncthreads();
    int acc = base[t];
    for (int i = t * 50; i < t * 50 + 50; i++) { off[i] = acc; acc += deg[i]; }
}

// Fill CSR (src lists per dst), sorted by source row -> fully deterministic.
__global__ void csr_fill_kernel(const float* __restrict__ adjSym, const int* __restrict__ hiBin,
                                const int* __restrict__ loBin, const int* __restrict__ tieNeed,
                                const int* __restrict__ tieCnt, const int* __restrict__ tieIdx,
                                const int* __restrict__ off, int* __restrict__ csrSrc)
{
    int d = blockIdx.x * blockDim.x + threadIdx.x;
    if (d >= NT) return;
    int b = d / NN, col = d % NN;
    unsigned thr = ((unsigned)hiBin[b] << 16) | (unsigned)loBin[b];
    const float* Ab = adjSym + (size_t)b * ADJN;
    const int* tix = tieIdx + b * TIECAP;
    int take = tieNeed[b], tc = tieCnt[b];
    int pos = off[d];
    for (int r = 0; r < NN; r++) {
        int flat = r * NN + col;
        if (edge_selected(__float_as_uint(Ab[flat]), thr, flat, tix, take, tc))
            csrSrc[pos++] = b * NN + r;
    }
}

// Layer-1 linear (K=16): one thread per output element.
__global__ void lin16_kernel(const float* __restrict__ x0, const float* __restrict__ W,
                             const float* __restrict__ bias, float* __restrict__ out, int N)
{
    int idx = blockIdx.x * blockDim.x + threadIdx.x;
    if (idx >= NT * N) return;
    int node = idx / N, o = idx % N;
    float v = bias[o];
#pragma unroll
    for (int k = 0; k < 16; k++) v += x0[node * 16 + k] * W[k * N + o];
    out[idx] = v;
}

// ---------------------------------------------------------------------------
// GATv2 edge softmax + aggregation. One wave per (dst-node, head).
// Two passes over CSR neighbors; logits cached in LDS (in-degree <= 200).
// SL: mask src==d edges and add one self loop (layers 2/3 semantics).
// ---------------------------------------------------------------------------
template <int H, int C, bool SL, bool OUTBF>
__global__ void __launch_bounds__(128)
agg_kernel(const float* __restrict__ xl, const float* __restrict__ xr,
           const float* __restrict__ att, const float* __restrict__ bias,
           const int* __restrict__ off, const int* __restrict__ csr,
           __bf16* __restrict__ obf, float* __restrict__ of32)
{
    constexpr int CPL = C / 32;
    constexpr int STR = H * C;
    __shared__ float slog[4][256];
    int wib  = threadIdx.x >> 5;
    int lane = threadIdx.x & 31;
    int task = blockIdx.x * 4 + wib;
    if (task >= NT * H) return;
    int d = task / H, h = task % H;

    const float* xrd = xr + (size_t)d * STR + h * C;
    float rx[CPL], at[CPL];
#pragma unroll
    for (int j = 0; j < CPL; j++) {
        rx[j] = xrd[lane + 32 * j];
        at[j] = att[h * C + lane + 32 * j];
    }
    int e0 = off[d], e1 = off[d + 1];
    float m = -INFINITY;
    for (int e = e0; e < e1; e++) {
        int s = csr[e];
        if (SL && s == d) { slog[wib][e - e0] = -INFINITY; continue; }
        const float* xls = xl + (size_t)s * STR + h * C;
        float p = 0.f;
#pragma unroll
        for (int j = 0; j < CPL; j++) {
            float v = xls[lane + 32 * j] + rx[j];
            v = v > 0.f ? v : 0.2f * v;            // leaky_relu(0.2)
            p += v * at[j];
        }
        for (int o = 16; o; o >>= 1) p += __shfl_xor(p, o, 32);
        slog[wib][e - e0] = p;
        m = fmaxf(m, p);
    }
    float selfLogit = 0.f;
    if (SL) {
        const float* xls = xl + (size_t)d * STR + h * C;
        float p = 0.f;
#pragma unroll
        for (int j = 0; j < CPL; j++) {
            float v = xls[lane + 32 * j] + rx[j];
            v = v > 0.f ? v : 0.2f * v;
            p += v * at[j];
        }
        for (int o = 16; o; o >>= 1) p += __shfl_xor(p, o, 32);
        selfLogit = p;
        m = fmaxf(m, p);
    }
    float denom = 0.f;
    float acc[CPL];
#pragma unroll
    for (int j = 0; j < CPL; j++) acc[j] = 0.f;
    for (int e = e0; e < e1; e++) {
        int s = csr[e];
        if (SL && s == d) continue;
        float ex = expf(slog[wib][e - e0] - m);
        denom += ex;
        const float* xls = xl + (size_t)s * STR + h * C;
#pragma unroll
        for (int j = 0; j < CPL; j++) acc[j] += ex * xls[lane + 32 * j];
    }
    if (SL) {
        float ex = expf(selfLogit - m);
        denom += ex;
        const float* xls = xl + (size_t)d * STR + h * C;
#pragma unroll
        for (int j = 0; j < CPL; j++) acc[j] += ex * xls[lane + 32 * j];
    }
    float inv = 1.f / (denom + 1e-16f);
#pragma unroll
    for (int j = 0; j < CPL; j++) {
        float v = geluf(acc[j] * inv + bias[h * C + lane + 32 * j]);
        size_t oidx = (size_t)d * STR + h * C + lane + 32 * j;
        if (OUTBF) obf[oidx] = f2bf(v);
        else       of32[oidx] = v;
    }
}

// global_mean_pool over 200 nodes per graph.
__global__ void pool_kernel(const float* __restrict__ h3, float* __restrict__ out)
{
    int idx = blockIdx.x * blockDim.x + threadIdx.x;
    if (idx >= BB * 512) return;
    int b = idx / 512, c = idx % 512;
    float s = 0.f;
    for (int n = 0; n < NN; n++) s += h3[((size_t)b * NN + n) * 512 + c];
    out[idx] = s * (1.f / (float)NN);
}

// ---------------------------------------------------------------------------
#define CEILDIV(a, b) (((a) + (b) - 1) / (b))

extern "C" void kernel_launch(void* const* d_in, const int* in_sizes, int n_in,
                              void* d_out, int out_size, void* d_ws, size_t ws_size,
                              hipStream_t stream)
{
    (void)in_sizes; (void)n_in; (void)out_size; (void)ws_size;

    const float* raw_fc = (const float*)d_in[0];
    const float* gbW    = (const float*)d_in[1];
    const float* gbb    = (const float*)d_in[2];
    const float* feW1   = (const float*)d_in[3];
    const float* feb1   = (const float*)d_in[4];
    const float* feW2   = (const float*)d_in[5];
    const float* feb2   = (const float*)d_in[6];
    const float* ln_g   = (const float*)d_in[7];
    const float* ln_b   = (const float*)d_in[8];
    const float* W1l = (const float*)d_in[9];  const float* b1l = (const float*)d_in[10];
    const float* W1r = (const float*)d_in[11]; const float* b1r = (const float*)d_in[12];
    const float* att1 = (const float*)d_in[13]; const float* bias1 = (const float*)d_in[14];
    const float* W2l = (const float*)d_in[15]; const float* b2l = (const float*)d_in[16];
    const float* W2r = (const float*)d_in[17]; const float* b2r = (const float*)d_in[18];
    const float* att2 = (const float*)d_in[19]; const float* bias2 = (const float*)d_in[20];
    const float* W3l = (const float*)d_in[21]; const float* b3l = (const float*)d_in[22];
    const float* W3r = (const float*)d_in[23]; const float* b3r = (const float*)d_in[24];
    const float* att3 = (const float*)d_in[25]; const float* bias3 = (const float*)d_in[26];
    float* out = (float*)d_out;

    // ---- workspace carving ----
    char* w = (char*)d_ws;
    auto alloc = [&](size_t bytes) -> char* {
        char* p = w; w += (bytes + 255) & ~(size_t)255; return p;
    };
    float*    adjRaw = (float*)alloc((size_t)BB * ADJN * 4);
    float*    adjSym = (float*)alloc((size_t)BB * ADJN * 4);
    __bf16*   rawbf  = (__bf16*)alloc((size_t)BB * DIN * 2);
    unsigned* hist   = (unsigned*)alloc((size_t)BB * 65536 * 4);
    int* hiBin   = (int*)alloc(BB * 4);
    int* hiNeed  = (int*)alloc(BB * 4);
    int* loBin   = (int*)alloc(BB * 4);
    int* tieNeed = (int*)alloc(BB * 4);
    int* tieCnt  = (int*)alloc(BB * 4);
    int* tieIdx  = (int*)alloc(BB * TIECAP * 4);
    float* means = (float*)alloc(NT * 4);
    float* stds  = (float*)alloc(NT * 4);
    float* x0    = (float*)alloc((size_t)NT * 16 * 4);
    int* deg     = (int*)alloc(NT * 4);
    int* off     = (int*)alloc((NT + 1) * 4);
    int* csrSrc  = (int*)alloc((size_t)BB * EPER * 4);
    float*  xlb  = (float*)alloc((size_t)NT * 1024 * 4);
    float*  xrb  = (float*)alloc((size_t)NT * 1024 * 4);
    __bf16* hbf  = (__bf16*)alloc((size_t)NT * 1024 * 2);
    float*  h3   = (float*)alloc((size_t)NT * 512 * 4);

    // ---- Stage A: adj = sigmoid(raw_fc @ gbW + gbb), bf16 WMMA (LDS tiled) ----
    cvt_bf16_kernel<<<CEILDIV(BB * DIN, 256), 256, 0, stream>>>(raw_fc, rawbf, BB * DIN);
    {
        int grid = (BB / 64) * CEILDIV(ADJN, 128);     // 1 * 313 (keep grid wide)
        wmma_gemm_bf16_kernel<2, ADJN, DIN, 1><<<grid, 256, 0, stream>>>(
            rawbf, gbW, gbb, adjRaw, BB);
    }

    // ---- Stage B: symmetrize + row stats + node-feature MLP + LayerNorm ----
    sym_stats_kernel<<<CEILDIV(NT * 32, 256), 256, 0, stream>>>(adjRaw, adjSym, means, stds);
    mlp_ln_kernel<<<CEILDIV(NT, 256), 256, 0, stream>>>(means, stds, feW1, feb1, feW2, feb2,
                                                        ln_g, ln_b, x0);

    // ---- Stage C: exact top-10000 per batch (two-level radix histogram) ----
    hipMemsetAsync(hist, 0, (size_t)BB * 65536 * 4, stream);
    hist_hi_kernel<<<CEILDIV(BB * ADJN, 256), 256, 0, stream>>>(adjSym, hist);
    hist_scan_kernel<<<BB, 256, 0, stream>>>(hist, nullptr, EPER, hiBin, hiNeed);
    hipMemsetAsync(hist, 0, (size_t)BB * 65536 * 4, stream);
    hist_lo_kernel<<<CEILDIV(BB * ADJN, 256), 256, 0, stream>>>(adjSym, hiBin, hist);
    hist_scan_kernel<<<BB, 256, 0, stream>>>(hist, hiNeed, 0, loBin, tieNeed);
    tie_scan_kernel<<<1, BB, 0, stream>>>(adjSym, hiBin, loBin, tieIdx, tieCnt);

    // ---- Stage D: CSR by destination (deterministic, sorted by src row) ----
    deg_kernel<<<CEILDIV(NT, 256), 256, 0, stream>>>(adjSym, hiBin, loBin, tieNeed, tieCnt,
                                                     tieIdx, deg);
    prefix_kernel<<<1, 256, 0, stream>>>(deg, off);
    csr_fill_kernel<<<CEILDIV(NT, 256), 256, 0, stream>>>(adjSym, hiBin, loBin, tieNeed, tieCnt,
                                                          tieIdx, off, csrSrc);

    // ---- Layer 1: GATv2(16 -> 8x128), concat ----
    lin16_kernel<<<CEILDIV(NT * 1024, 256), 256, 0, stream>>>(x0, W1l, b1l, xlb, 1024);
    lin16_kernel<<<CEILDIV(NT * 1024, 256), 256, 0, stream>>>(x0, W1r, b1r, xrb, 1024);
    agg_kernel<8, 128, false, true><<<CEILDIV(NT * 8, 4), 128, 0, stream>>>(
        xlb, xrb, att1, bias1, off, csrSrc, hbf, nullptr);

    // ---- Layer 2: GATv2(1024 -> 4x256), self loops, concat ----
    {
        int grid = (NT / 64) * CEILDIV(1024, 256);     // 200 * 4, wide 8-wmma tiles
        wmma_gemm_bf16_kernel<4, 1024, 1024, 0><<<grid, 256, 0, stream>>>(
            hbf, W2l, b2l, xlb, NT);
        wmma_gemm_bf16_kernel<4, 1024, 1024, 0><<<grid, 256, 0, stream>>>(
            hbf, W2r, b2r, xrb, NT);
    }
    agg_kernel<4, 256, true, true><<<CEILDIV(NT * 4, 4), 128, 0, stream>>>(
        xlb, xrb, att2, bias2, off, csrSrc, hbf, nullptr);

    // ---- Layer 3: GATv2(1024 -> 1x512), self loops, mean heads ----
    {
        int grid = (NT / 64) * CEILDIV(512, 256);      // 200 * 2
        wmma_gemm_bf16_kernel<4, 512, 1024, 0><<<grid, 256, 0, stream>>>(
            hbf, W3l, b3l, xlb, NT);
        wmma_gemm_bf16_kernel<4, 512, 1024, 0><<<grid, 256, 0, stream>>>(
            hbf, W3r, b3r, xrb, NT);
    }
    agg_kernel<1, 512, true, false><<<CEILDIV(NT * 1, 4), 128, 0, stream>>>(
        xlb, xrb, att3, bias3, off, csrSrc, nullptr, h3);

    // ---- global_mean_pool ----
    pool_kernel<<<CEILDIV(BB * 512, 256), 256, 0, stream>>>(h3, out);
}
